// LinearAttention_14345190769334
// MI455X (gfx1250) — matrix operand order repaired and checked by reading, not compiled, not run on hardware
//
#include <hip/hip_runtime.h>
#include <math.h>

// ---------------------------------------------------------------------------
// LinearAttention on gfx1250 (wave32), all GEMMs via v_wmma_f32_16x16x32_f16.
// 2x2 register blocking (32x32 output per wave): every fragment load is two
// contiguous global_load_b128 per lane, and each fragment feeds 2 WMMAs
// (-> 2 b128 loads per WMMA).
// ---------------------------------------------------------------------------

typedef __attribute__((ext_vector_type(16))) _Float16 v16h;
typedef __attribute__((ext_vector_type(8)))  _Float16 v8h;
typedef __attribute__((ext_vector_type(8)))  float    v8f;

#define NSP   4096   // N = H*W
#define CDIM  256
#define HID   128
#define NB    16     // batches

union HFrag {
    v16h v;
    _Float16 h[16];
    uint4 q[2];
};

__device__ __forceinline__ v8f wmma_f16(v16h a, v16h b, v8f c) {
    return __builtin_amdgcn_wmma_f32_16x16x32_f16(
        /*neg_a=*/false, a, /*neg_b=*/false, b,
        /*c_mod=*/(short)0, c, /*reuse_a=*/false, /*reuse_b=*/false);
}

// A fragment (16x32 tile of row-major MxK at (row0,k0)).
// ISA 7.12.2: lane(g,m): e<8 -> K=k0+8g+e ; e>=8 -> K=k0+16+8g+(e-8).
__device__ __forceinline__ v16h load_a_rowmajor(const _Float16* __restrict__ A,
                                                int ld, int row0, int k0,
                                                int g, int m) {
    HFrag u;
    const _Float16* p = A + (size_t)(row0 + m) * ld + k0 + 8 * g;
    u.q[0] = *(const uint4*)(p);
    u.q[1] = *(const uint4*)(p + 16);
    return u.v;
}

// B fragment (32x16 tile at (k0,n0)) read from B^T stored row-major
// (T[n][k] = B[k][n]): lane(g,n): e<8 -> K=k0+8g+e ; e>=8 -> K=k0+16+8g+(e-8).
__device__ __forceinline__ v16h load_b_from_T(const _Float16* __restrict__ T,
                                              int ld, int k0, int n0,
                                              int g, int m) {
    HFrag u;
    const _Float16* p = T + (size_t)(n0 + m) * ld + k0 + 8 * g;
    u.q[0] = *(const uint4*)(p);
    u.q[1] = *(const uint4*)(p + 16);
    return u.v;
}

// Transposed D store: lane (g,m) holds D[M=8g+r][N=m]; in an N-major dest
// those 8 values are contiguous -> one 16-byte store.
__device__ __forceinline__ void store_d_T(_Float16* __restrict__ dst, v8f acc) {
    v8h o;
#pragma unroll
    for (int r = 0; r < 8; ++r) o[r] = (_Float16)acc[r];
    *(v8h*)dst = o;
}

// ---------------------------------------------------------------------------
// Stage 0a: x[b][c][n] (f32) -> xt[b][n][c] (f16), LDS-tiled 32x32 transpose.
__global__ void k_cvt_xT(const float* __restrict__ x, _Float16* __restrict__ xt) {
    __shared__ float tile[32][33];
    int blk   = blockIdx.x;
    int ntile = blk & 127;           // 4096/32
    int ctile = (blk >> 7) & 7;      // 256/32
    int b     = blk >> 10;
    int tid = threadIdx.x;
    int col = tid & 31;
    int row = tid >> 5;              // 0..7
    const float* xb = x + (size_t)b * CDIM * NSP;
#pragma unroll
    for (int it = 0; it < 4; ++it) {
        int c = ctile * 32 + row + it * 8;
        tile[row + it * 8][col] = xb[(size_t)c * NSP + ntile * 32 + col];
    }
    __syncthreads();
    _Float16* xtb = xt + (size_t)b * NSP * CDIM;
#pragma unroll
    for (int it = 0; it < 4; ++it) {
        int n = ntile * 32 + row + it * 8;
        xtb[(size_t)n * CDIM + ctile * 32 + col] = (_Float16)tile[col][row + it * 8];
    }
}

// Stage 0b: pack Wq/Wk/Wv -> [384][256] f16; Wout -> [256][128] f16.
__global__ void k_cvt_w(const float* __restrict__ Wq, const float* __restrict__ Wk,
                        const float* __restrict__ Wv, const float* __restrict__ Wout,
                        _Float16* __restrict__ wqkv, _Float16* __restrict__ wout) {
    int i = blockIdx.x * blockDim.x + threadIdx.x;     // 0 .. 131071
    if (i < 3 * HID * CDIM) {
        int sel = i / (HID * CDIM);
        int idx = i - sel * (HID * CDIM);
        const float* src = (sel == 0) ? Wq : (sel == 1) ? Wk : Wv;
        wqkv[i] = (_Float16)src[idx];
    } else {
        int j = i - 3 * HID * CDIM;
        wout[j] = (_Float16)Wout[j];
    }
}

// Stage 0c: zero the f32 context accumulator [B][4][32][32].
__global__ void k_zero_ctx(float* __restrict__ ctx) {
    ctx[blockIdx.x * blockDim.x + threadIdx.x] = 0.0f;
}

// ---------------------------------------------------------------------------
// Stage 1: fused q/k/v projections, 32x32 output per wave (2x2 WMMA tiles).
// o-groups 0..3 -> qT[b][n][128]; o-groups 4..11 -> kv[b][o-128][n].
__global__ void k_gemm_qkv(const _Float16* __restrict__ xt,
                           const _Float16* __restrict__ wqkv,
                           _Float16* __restrict__ qt,
                           _Float16* __restrict__ kv) {
    int wave = (blockIdx.x * blockDim.x + threadIdx.x) >> 5;
    int lane = threadIdx.x & 31, g = lane >> 4, m = lane & 15;
    int ng = wave & 127;                 // 128 n-groups of 32
    int og = (wave >> 7) % 12;           // 12 o-groups of 32 (384)
    int b  = wave / (128 * 12);
    int n0 = ng * 32, o0 = og * 32;
    const _Float16* xtb = xt + (size_t)b * NSP * CDIM;

    v8f a00 = {}, a01 = {}, a10 = {}, a11 = {};
#pragma unroll
    for (int k = 0; k < CDIM; k += 32) {
        v16h wa0 = load_a_rowmajor(wqkv, CDIM, o0,      k, g, m);
        v16h wa1 = load_a_rowmajor(wqkv, CDIM, o0 + 16, k, g, m);
        v16h xb0 = load_b_from_T(xtb, CDIM, k, n0,      g, m);
        v16h xb1 = load_b_from_T(xtb, CDIM, k, n0 + 16, g, m);
        a00 = wmma_f16(wa0, xb0, a00);
        a01 = wmma_f16(wa0, xb1, a01);
        a10 = wmma_f16(wa1, xb0, a10);
        a11 = wmma_f16(wa1, xb1, a11);
    }
    v8f acc[2][2] = {{a00, a01}, {a10, a11}};
    if (og < 4) {
        // q block: transposed, one 16B store per lane per subtile.
#pragma unroll
        for (int sm = 0; sm < 2; ++sm)
#pragma unroll
            for (int sn = 0; sn < 2; ++sn) {
                _Float16* dst = qt + ((size_t)b * NSP + n0 + sn * 16 + m) * HID
                                + o0 + sm * 16 + 8 * g;
                store_d_T(dst, acc[sm][sn]);
            }
    } else {
        _Float16* dst = kv + (size_t)b * 256 * NSP;
#pragma unroll
        for (int sm = 0; sm < 2; ++sm)
#pragma unroll
            for (int sn = 0; sn < 2; ++sn)
#pragma unroll
                for (int r = 0; r < 8; ++r) {
                    int row = o0 - 128 + sm * 16 + r + 8 * g;
                    dst[(size_t)row * NSP + n0 + sn * 16 + m] =
                        (_Float16)acc[sm][sn][r];
                }
    }
}

// ---------------------------------------------------------------------------
// Stage 2: in-place softmax over n for k rows (kv rows 0..127 per batch).
__global__ void k_softmax(_Float16* __restrict__ kv) {
    __shared__ float sm[256];
    int row = blockIdx.x;                 // 0 .. B*128-1
    int b = row >> 7, d = row & 127;
    _Float16* p = kv + ((size_t)b * 256 + d) * NSP;
    int tid = threadIdx.x;

    float vals[16];
    float mx = -3.0e38f;
#pragma unroll
    for (int t = 0; t < 16; ++t) {
        vals[t] = (float)p[tid + t * 256];
        mx = fmaxf(mx, vals[t]);
    }
    sm[tid] = mx; __syncthreads();
    for (int s = 128; s > 0; s >>= 1) {
        if (tid < s) sm[tid] = fmaxf(sm[tid], sm[tid + s]);
        __syncthreads();
    }
    mx = sm[0]; __syncthreads();

    float sum = 0.0f;
#pragma unroll
    for (int t = 0; t < 16; ++t) { vals[t] = __expf(vals[t] - mx); sum += vals[t]; }
    sm[tid] = sum; __syncthreads();
    for (int s = 128; s > 0; s >>= 1) {
        if (tid < s) sm[tid] += sm[tid + s];
        __syncthreads();
    }
    float inv = 1.0f / sm[0];
#pragma unroll
    for (int t = 0; t < 16; ++t) p[tid + t * 256] = (_Float16)(vals[t] * inv);
}

// ---------------------------------------------------------------------------
// Stage 3: context[b,h][d,e] = sum_n ksm[d,n]*v[e,n].  Full 32x32 per wave
// (2x2 WMMA), K=4096 split 64 ways, f32 atomics into zeroed ctx.
__global__ void k_context(const _Float16* __restrict__ kv, float* __restrict__ ctx) {
    int wave = (blockIdx.x * blockDim.x + threadIdx.x) >> 5;
    int lane = threadIdx.x & 31, g = lane >> 4, m = lane & 15;
    int ks = wave & 63;           // 64-way K split (chunk = 64)
    int h  = (wave >> 6) & 3;
    int b  = wave >> 8;
    const _Float16* base = kv + (size_t)b * 256 * NSP;
    const int ar0 = h * 32;             // ksm rows (kv 0..127)
    const _Float16* v0 = base + (size_t)(128 + h * 32) * NSP;       // v rows
    const _Float16* v1 = base + (size_t)(128 + h * 32 + 16) * NSP;

    v8f a00 = {}, a01 = {}, a10 = {}, a11 = {};
#pragma unroll
    for (int kk = 0; kk < 64; kk += 32) {
        int k = ks * 64 + kk;
        v16h ka0 = load_a_rowmajor(base, NSP, ar0,      k, g, m);
        v16h ka1 = load_a_rowmajor(base, NSP, ar0 + 16, k, g, m);
        v16h vb0 = load_b_from_T(v0, NSP, k, 0, g, m);
        v16h vb1 = load_b_from_T(v1, NSP, k, 0, g, m);
        a00 = wmma_f16(ka0, vb0, a00);
        a01 = wmma_f16(ka0, vb1, a01);
        a10 = wmma_f16(ka1, vb0, a10);
        a11 = wmma_f16(ka1, vb1, a11);
    }
    v8f acc[2][2] = {{a00, a01}, {a10, a11}};
    float* c = ctx + (size_t)(b * 4 + h) * 32 * 32;
#pragma unroll
    for (int sm = 0; sm < 2; ++sm)
#pragma unroll
        for (int sn = 0; sn < 2; ++sn)
#pragma unroll
            for (int r = 0; r < 8; ++r) {
                int d = sm * 16 + r + 8 * g;
                atomicAdd(&c[d * 32 + sn * 16 + m], acc[sm][sn][r]);
            }
}

// ---------------------------------------------------------------------------
// Stage 4: out[e,n] = sum_d ctx[d][e]*q[d,n] -> outT[b][n][128].
// Per wave: both e-tiles x 2 n-tiles (4 WMMA, single K step of 32).
__global__ void k_outstage(const _Float16* __restrict__ qt,
                           const float* __restrict__ ctx,
                           _Float16* __restrict__ outt) {
    int wave = (blockIdx.x * blockDim.x + threadIdx.x) >> 5;
    int lane = threadIdx.x & 31, g = lane >> 4, m = lane & 15;
    int ng = wave & 127;          // 128 n-groups of 32
    int h  = (wave >> 7) & 3;
    int b  = wave >> 9;
    int n0 = ng * 32;
    const _Float16* qtb = qt + (size_t)b * NSP * HID;
    const float* c = ctx + (size_t)(b * 4 + h) * 32 * 32;

    // A[e][d] = ctx[d][e] for e-tile 0 and 1, converted f32->f16.
    v16h ae[2];
#pragma unroll
    for (int se = 0; se < 2; ++se) {
        HFrag ua;
        int erow = se * 16 + m;
#pragma unroll
        for (int el = 0; el < 16; ++el) {
            int d = (el < 8) ? (el + 8 * g) : (16 + (el - 8) + 8 * g);
            ua.h[el] = (_Float16)c[d * 32 + erow];
        }
        ae[se] = ua.v;
    }
    v16h qb0 = load_b_from_T(qtb, HID, h * 32, n0,      g, m);
    v16h qb1 = load_b_from_T(qtb, HID, h * 32, n0 + 16, g, m);

    v8f acc[2][2] = {};
#pragma unroll
    for (int se = 0; se < 2; ++se) {
        acc[se][0] = wmma_f16(ae[se], qb0, acc[se][0]);
        acc[se][1] = wmma_f16(ae[se], qb1, acc[se][1]);
    }
#pragma unroll
    for (int se = 0; se < 2; ++se)
#pragma unroll
        for (int sn = 0; sn < 2; ++sn) {
            _Float16* dst = outt + ((size_t)b * NSP + n0 + sn * 16 + m) * HID
                            + h * 32 + se * 16 + 8 * g;
            store_d_T(dst, acc[se][sn]);
        }
}

// ---------------------------------------------------------------------------
// Stage 5: y[b][c][n] = sum_o Wout[c][o]*out[o][n] + bias[c] (f32 out).
// 32x32 output per wave (2x2 WMMA), K=128 -> 4 steps.
__global__ void k_final(const _Float16* __restrict__ outt,
                        const _Float16* __restrict__ wout,
                        const float* __restrict__ bias,
                        float* __restrict__ y) {
    int wave = (blockIdx.x * blockDim.x + threadIdx.x) >> 5;
    int lane = threadIdx.x & 31, g = lane >> 4, m = lane & 15;
    int ng = wave & 127;          // 128 n-groups of 32
    int cg = (wave >> 7) & 7;     // 8 c-groups of 32 (256)
    int b  = wave >> 10;
    int n0 = ng * 32, c0 = cg * 32;
    const _Float16* ob = outt + (size_t)b * NSP * HID;

    v8f a00 = {}, a01 = {}, a10 = {}, a11 = {};
#pragma unroll
    for (int k = 0; k < HID; k += 32) {
        v16h wa0 = load_a_rowmajor(wout, HID, c0,      k, g, m);
        v16h wa1 = load_a_rowmajor(wout, HID, c0 + 16, k, g, m);
        v16h bb0 = load_b_from_T(ob, HID, k, n0,      g, m);
        v16h bb1 = load_b_from_T(ob, HID, k, n0 + 16, g, m);
        a00 = wmma_f16(wa0, bb0, a00);
        a01 = wmma_f16(wa0, bb1, a01);
        a10 = wmma_f16(wa1, bb0, a10);
        a11 = wmma_f16(wa1, bb1, a11);
    }
    v8f acc[2][2] = {{a00, a01}, {a10, a11}};
    float* yb = y + (size_t)b * CDIM * NSP;
#pragma unroll
    for (int sm = 0; sm < 2; ++sm)
#pragma unroll
        for (int sn = 0; sn < 2; ++sn)
#pragma unroll
            for (int r = 0; r < 8; ++r) {
                int cc = c0 + sm * 16 + r + 8 * g;
                yb[(size_t)cc * NSP + n0 + sn * 16 + m] =
                    acc[sm][sn][r] + bias[cc];
            }
}

// ---------------------------------------------------------------------------
extern "C" void kernel_launch(void* const* d_in, const int* in_sizes, int n_in,
                              void* d_out, int out_size, void* d_ws, size_t ws_size,
                              hipStream_t stream) {
    const float* x    = (const float*)d_in[0];
    const float* Wq   = (const float*)d_in[1];
    const float* Wk   = (const float*)d_in[2];
    const float* Wv   = (const float*)d_in[3];
    const float* Wout = (const float*)d_in[4];
    const float* bout = (const float*)d_in[5];
    float* y = (float*)d_out;

    // Workspace layout (bytes), ~96.5 MB total, all regions 16B-aligned.
    char* ws = (char*)d_ws;
    const size_t OFF_XT   = 0;                                        // 32 MB
    const size_t OFF_WQKV = OFF_XT   + (size_t)NB * NSP * CDIM * 2;
    const size_t OFF_WOUT = OFF_WQKV + (size_t)384 * CDIM * 2;
    const size_t OFF_QT   = OFF_WOUT + (size_t)CDIM * HID * 2;        // 16 MB
    const size_t OFF_KV   = OFF_QT   + (size_t)NB * NSP * HID * 2;    // 32 MB
    const size_t OFF_OUTT = OFF_KV   + (size_t)NB * 256 * NSP * 2;    // 16 MB
    const size_t OFF_CTX  = OFF_OUTT + (size_t)NB * NSP * HID * 2;
    _Float16* xt   = (_Float16*)(ws + OFF_XT);
    _Float16* wqkv = (_Float16*)(ws + OFF_WQKV);
    _Float16* wout = (_Float16*)(ws + OFF_WOUT);
    _Float16* qt   = (_Float16*)(ws + OFF_QT);
    _Float16* kv   = (_Float16*)(ws + OFF_KV);
    _Float16* outt = (_Float16*)(ws + OFF_OUTT);
    float*    ctx  = (float*)   (ws + OFF_CTX);

    // Stage 0: transpose-convert x, convert weights, zero ctx.
    k_cvt_xT  <<<16384, 256, 0, stream>>>(x, xt);
    k_cvt_w   <<<512,   256, 0, stream>>>(Wq, Wk, Wv, Wout, wqkv, wout);
    k_zero_ctx<<<256,   256, 0, stream>>>(ctx);

    // Stage 1: fused q/k/v projections, 24576 waves (32x32 tiles).
    k_gemm_qkv<<<3072,  256, 0, stream>>>(xt, wqkv, qt, kv);
    // Stage 2: softmax over n on k rows (in place), 2048 rows.
    k_softmax <<<2048,  256, 0, stream>>>(kv);
    // Stage 3: context, 64-way K split, 4096 waves.
    k_context <<<512,   256, 0, stream>>>(kv, ctx);
    // Stage 4: out = ctx^T * q -> outT, 8192 waves.
    k_outstage<<<1024,  256, 0, stream>>>(qt, ctx, outt);
    // Stage 5: final projection + bias -> f32 d_out, 16384 waves.
    k_final   <<<2048,  256, 0, stream>>>(outt, wout, bout, y);
}
